// Struct2Seq_51548197486893
// MI455X (gfx1250) — matrix operand order repaired and checked by reading, not compiled, not run on hardware
//
#include <hip/hip_runtime.h>
#include <cstdint>
#include <cstddef>

// ---------------------------------------------------------------------------
// Types and bf16 helpers
// ---------------------------------------------------------------------------
typedef __bf16 bf16_t;
typedef __attribute__((ext_vector_type(16))) __bf16 v16bf;
typedef __attribute__((ext_vector_type(2)))  __bf16 v2bf;
typedef __attribute__((ext_vector_type(8)))  float  v8f;

#if defined(__has_builtin) && __has_builtin(__builtin_amdgcn_cvt_pk_bf16_f32)
__device__ __forceinline__ unsigned pack2f(float lo, float hi) {
  v2bf r = __builtin_amdgcn_cvt_pk_bf16_f32(lo, hi);
  return __builtin_bit_cast(unsigned, r);
}
__device__ __forceinline__ bf16_t f2bf(float f) {
  v2bf r = __builtin_amdgcn_cvt_pk_bf16_f32(f, 0.f);
  return r[0];
}
#else
__device__ __forceinline__ unsigned f2bfbits(float f) {
  unsigned u = __builtin_bit_cast(unsigned, f);
  return (u + 0x7FFFu + ((u >> 16) & 1u)) >> 16;   // round-to-nearest-even
}
__device__ __forceinline__ unsigned pack2f(float lo, float hi) {
  return f2bfbits(lo) | (f2bfbits(hi) << 16);
}
__device__ __forceinline__ bf16_t f2bf(float f) {
  return __builtin_bit_cast(bf16_t, (unsigned short)f2bfbits(f));
}
#endif

__device__ __forceinline__ float bflo(unsigned u) { return __builtin_bit_cast(float, u << 16); }
__device__ __forceinline__ float bfhi(unsigned u) { return __builtin_bit_cast(float, u & 0xFFFF0000u); }

__device__ __forceinline__ v8f wmma_bf16(v16bf a, v16bf b, v8f c) {
  return __builtin_amdgcn_wmma_f32_16x16x32_bf16(false, a, false, b, (short)0, c, false, false);
}

// LDS row stride padding: +8 bf16 (16B) keeps b128 alignment while making the
// row stride == 4 dwords (mod 64 banks) -> the 16 lanes of an A-fragment
// ds_load_b128 land on disjoint 4-bank groups (conflict-free).
constexpr int kYLD = 136;   // padded stride for 128-wide LDS tiles

// ---------------------------------------------------------------------------
// Fragment loaders
//
// A fragment (16x32 bf16, MxK): lane l holds row m=(l&15); vector elems 0..7
// are K = kb + (l>>4)*8 + 0..7, elems 8..15 are K = kb + 16 + (l>>4)*8 + 0..7.
// B fragment: pre-packed so each lane reads 16 contiguous bf16 (32B).
// C/D (16x16 f32): lane l col n=(l&15), VGPR v row m = v + 8*(l>>4).
// ---------------------------------------------------------------------------
__device__ __forceinline__ v16bf lds_a_frag(const bf16_t* Xs, int ld, int mt, int kb) {
  int l = threadIdx.x & 31;
  int m = mt * 16 + (l & 15);
  int hh = (l >> 4) * 8;
  const bf16_t* p = Xs + m * ld + kb + hh;
  union { v16bf v; uint4 q[2]; } U;
  U.q[0] = *(const uint4*)(p);
  U.q[1] = *(const uint4*)(p + 16);
  return U.v;
}

template <typename TA>
__device__ __forceinline__ v16bf glb_a_frag(const TA* A, int ld, int mbase, int mt, int kb) {
  int l = threadIdx.x & 31;
  int m = mbase + mt * 16 + (l & 15);
  int hh = (l >> 4) * 8;
  const TA* p = A + (size_t)m * ld + kb + hh;
  if constexpr (sizeof(TA) == 4) {          // f32 source: packed convert
    union { v16bf v; unsigned u[8]; } U;
#pragma unroll
    for (int e = 0; e < 4; ++e) U.u[e] = pack2f(p[2 * e], p[2 * e + 1]);
#pragma unroll
    for (int e = 0; e < 4; ++e) U.u[4 + e] = pack2f(p[16 + 2 * e], p[16 + 2 * e + 1]);
    return U.v;
  } else {                                   // bf16 source: two 16B loads
    union { v16bf v; uint4 q[2]; } U;
    U.q[0] = *(const uint4*)(p);
    U.q[1] = *(const uint4*)(p + 16);
    return U.v;
  }
}

__device__ __forceinline__ v16bf b_frag(const bf16_t* Wp, int n16, int kt, int nt) {
  int l = threadIdx.x & 31;
  const bf16_t* p = Wp + ((size_t)(kt * n16 + nt) * 32 + l) * 16;
  union { v16bf v; uint4 q[2]; } U;
  U.q[0] = *(const uint4*)(p);
  U.q[1] = *(const uint4*)(p + 8);
  return U.v;
}

// ---------------------------------------------------------------------------
// Weight packing: W (KxN f32, row-major) -> WMMA-B fragment order (bf16)
// element layout: [((kt*n16+nt)*32 + lane)*16 + 2v] = W[kt*32+(v>>2)*16+(l>>4)*8+2(v&3)][nt*16+(l&15)]
// ---------------------------------------------------------------------------
__global__ void pack_w_k(const float* __restrict__ W, bf16_t* __restrict__ Wp,
                         int K, int N, int Npad) {
  int n16 = Npad >> 4;
  int total = (K >> 5) * n16 * 256;
  for (int i = blockIdx.x * blockDim.x + threadIdx.x; i < total; i += gridDim.x * blockDim.x) {
    int v = i & 7;
    int l = (i >> 3) & 31;
    int tile = i >> 8;
    int ntt = tile % n16;
    int kt = tile / n16;
    int k = kt * 32 + (v >> 2) * 16 + (l >> 4) * 8 + 2 * (v & 3);
    int n = ntt * 16 + (l & 15);
    float lo = (n < N) ? W[(size_t)k * N + n] : 0.f;
    float hi = (n < N) ? W[(size_t)(k + 1) * N + n] : 0.f;
    *(unsigned*)(Wp + (size_t)i * 2) = pack2f(lo, hi);
  }
}

// ---------------------------------------------------------------------------
// Generic GEMM: C[M x Nd] = act(A[M x Kd] @ Wp + bias).  One wave per
// (32-row, 16-col) output tile.  WF/WB select f32 / bf16 outputs statically.
// ---------------------------------------------------------------------------
template <typename TA, int RELU, int WF, int WB>
__global__ __launch_bounds__(32) void gemm_k(const TA* __restrict__ A,
                                             const bf16_t* __restrict__ Wp,
                                             const float* __restrict__ bias,
                                             float* __restrict__ Cf,
                                             bf16_t* __restrict__ Cb,
                                             int Kd, int Nd) {
  int mb = blockIdx.x;
  int nt = blockIdx.y;
  int n16 = Nd >> 4;
  int l = threadIdx.x;
  v8f acc0 = {}, acc1 = {};
  for (int kt = 0; kt < (Kd >> 5); ++kt) {
    v16bf bfrag = b_frag(Wp, n16, kt, nt);
    v16bf a0 = glb_a_frag(A, Kd, mb * 32, 0, kt * 32);
    v16bf a1 = glb_a_frag(A, Kd, mb * 32, 1, kt * 32);
    acc0 = wmma_bf16(a0, bfrag, acc0);
    acc1 = wmma_bf16(a1, bfrag, acc1);
  }
  int h = l >> 4;
  int n = nt * 16 + (l & 15);
  float bs = bias[n];
#pragma unroll
  for (int v = 0; v < 8; ++v) {
    int m0 = mb * 32 + v + 8 * h;
    float x0 = acc0[v] + bs;
    if (RELU) x0 = fmaxf(x0, 0.f);
    if (WF) Cf[(size_t)m0 * Nd + n] = x0;
    if (WB) Cb[(size_t)m0 * Nd + n] = f2bf(x0);
    int m1 = m0 + 16;
    float x1 = acc1[v] + bs;
    if (RELU) x1 = fmaxf(x1, 0.f);
    if (WF) Cf[(size_t)m1 * Nd + n] = x1;
    if (WB) Cb[(size_t)m1 * Nd + n] = f2bf(x1);
  }
}

// ---------------------------------------------------------------------------
// In-workgroup GEMM stages for the fused edge-message kernel.
// 4 waves; wave w owns output columns [w*32, w*32+32).  Y is 32 x 128 with
// padded stride kYLD.  LDI = padded input row stride.
// ---------------------------------------------------------------------------
template <int KIN, int LDI>
__device__ __forceinline__ void mlp_gemm_relu(const bf16_t* Xs, const bf16_t* Wp,
                                              const float* bias, bf16_t* Yd) {
  int wave = threadIdx.x >> 5;
  int l = threadIdx.x & 31;
  v8f a00 = {}, a01 = {}, a10 = {}, a11 = {};
  for (int kt = 0; kt < (KIN >> 5); ++kt) {
    v16bf fa0 = lds_a_frag(Xs, LDI, 0, kt * 32);
    v16bf fa1 = lds_a_frag(Xs, LDI, 1, kt * 32);
    v16bf fb0 = b_frag(Wp, 8, kt, wave * 2 + 0);
    v16bf fb1 = b_frag(Wp, 8, kt, wave * 2 + 1);
    a00 = wmma_bf16(fa0, fb0, a00);
    a10 = wmma_bf16(fa1, fb0, a10);
    a01 = wmma_bf16(fa0, fb1, a01);
    a11 = wmma_bf16(fa1, fb1, a11);
  }
  int h = l >> 4;
#pragma unroll
  for (int nt = 0; nt < 2; ++nt) {
    int n = wave * 32 + nt * 16 + (l & 15);
    float bs = bias[n];
    const v8f& A0 = nt ? a01 : a00;
    const v8f& A1 = nt ? a11 : a10;
#pragma unroll
    for (int v = 0; v < 8; ++v) {
      int m0 = v + 8 * h;
      Yd[m0 * kYLD + n] = f2bf(fmaxf(A0[v] + bs, 0.f));
      Yd[(m0 + 16) * kYLD + n] = f2bf(fmaxf(A1[v] + bs, 0.f));
    }
  }
}

__device__ __forceinline__ void mlp_gemm_sum(const bf16_t* Xs, const bf16_t* Wp,
                                             const float* bias, const float* mk,
                                             float* colred, float* dhout) {
  int wave = threadIdx.x >> 5;
  int l = threadIdx.x & 31;
  v8f a00 = {}, a01 = {}, a10 = {}, a11 = {};
  for (int kt = 0; kt < 4; ++kt) {
    v16bf fa0 = lds_a_frag(Xs, kYLD, 0, kt * 32);
    v16bf fa1 = lds_a_frag(Xs, kYLD, 1, kt * 32);
    v16bf fb0 = b_frag(Wp, 8, kt, wave * 2 + 0);
    v16bf fb1 = b_frag(Wp, 8, kt, wave * 2 + 1);
    a00 = wmma_bf16(fa0, fb0, a00);
    a10 = wmma_bf16(fa1, fb0, a10);
    a01 = wmma_bf16(fa0, fb1, a01);
    a11 = wmma_bf16(fa1, fb1, a11);
  }
  int h = l >> 4;
#pragma unroll
  for (int nt = 0; nt < 2; ++nt) {
    int n = wave * 32 + nt * 16 + (l & 15);
    float bs = bias[n];
    const v8f& A0 = nt ? a01 : a00;
    const v8f& A1 = nt ? a11 : a10;
    float s = 0.f;
#pragma unroll
    for (int v = 0; v < 8; ++v) {
      int m0 = v + 8 * h;
      s += mk[m0] * (A0[v] + bs);
      s += mk[m0 + 16] * (A1[v] + bs);
    }
    colred[h * 128 + n] = s;
  }
  __syncthreads();
  int t = threadIdx.x;
  if (t < 128) dhout[t] = colred[t] + colred[128 + t];
}

// ---------------------------------------------------------------------------
// Fused edge-message kernel: one workgroup per node; builds the 32 x KD
// concatenated h_EV tile in LDS and runs the 3-layer edge MLP + masked K-sum.
// DEC=0: rows = [hV[n] | hE | hV[nbr]]                (KD=384)
// DEC=1: rows = [hV[n] | m*hE | mbw*hS[nbr] | mbw*hV[nbr]+mfw*hVenc[nbr]] (KD=512)
// ---------------------------------------------------------------------------
template <int DEC>
__global__ __launch_bounds__(128) void edge_msg_k(
    const bf16_t* __restrict__ hVb, const bf16_t* __restrict__ hEb,
    const int* __restrict__ E_idx, const float* __restrict__ mask,
    const bf16_t* __restrict__ hSb, const bf16_t* __restrict__ hVencb,
    const float* __restrict__ mask_bw,
    const bf16_t* __restrict__ W1p, const float* __restrict__ b1,
    const bf16_t* __restrict__ W2p, const float* __restrict__ b2,
    const bf16_t* __restrict__ W3p, const float* __restrict__ b3,
    float* __restrict__ dh, int N) {
  constexpr int KD = DEC ? 512 : 384;
  constexpr int XLD = KD + 8;           // padded LDS stride (bank-conflict-free)
  constexpr int KNB = 30;
  __shared__ alignas(16) bf16_t X[32 * XLD];
  __shared__ alignas(16) bf16_t Y1[32 * kYLD];
  __shared__ alignas(16) bf16_t Y2[32 * kYLD];
  __shared__ float colred[256];
  __shared__ int nbr[32];
  __shared__ float mkA[32];
  __shared__ float mbwL[32];

  int node = blockIdx.x;
  int b = node / N;
  int t = threadIdx.x;
  float mnode = mask[node];

  if (t < 32) {
    int r = t;
    int j = (r < KNB) ? E_idx[(size_t)node * KNB + r] : 0;
    nbr[r] = j;
    if (DEC) {
      mbwL[r] = (r < KNB) ? mask_bw[(size_t)node * KNB + r] : 0.f;
      mkA[r] = (r < KNB) ? 1.f : 0.f;
    } else {
      mbwL[r] = 0.f;
      mkA[r] = (r < KNB) ? mnode * mask[(size_t)b * N + j] : 0.f;
    }
  }
  __syncthreads();

  for (int i = t; i < 32 * KD / 2; i += 128) {
    int r = i / (KD / 2);
    int c = (i % (KD / 2)) * 2;
    unsigned val = 0u;
    if (r < KNB) {
      if (c < 128) {
        val = *(const unsigned*)(hVb + (size_t)node * 128 + c);
      } else if (c < 256) {
        unsigned e = *(const unsigned*)(hEb + ((size_t)node * KNB + r) * 128 + (c - 128));
        if (DEC) val = pack2f(mnode * bflo(e), mnode * bfhi(e));
        else     val = e;
      } else if (c < 384) {
        int j = nbr[r];
        if (!DEC) {
          val = *(const unsigned*)(hVb + ((size_t)b * N + j) * 128 + (c - 256));
        } else {
          unsigned s = *(const unsigned*)(hSb + ((size_t)b * N + j) * 128 + (c - 256));
          float mb = mbwL[r];
          val = pack2f(mb * bflo(s), mb * bfhi(s));
        }
      } else {  // DEC only
        int j = nbr[r];
        unsigned cu = *(const unsigned*)(hVb + ((size_t)b * N + j) * 128 + (c - 384));
        unsigned eu = *(const unsigned*)(hVencb + ((size_t)b * N + j) * 128 + (c - 384));
        float mb = mbwL[r], mf = mnode - mbwL[r];
        val = pack2f(mb * bflo(cu) + mf * bflo(eu), mb * bfhi(cu) + mf * bfhi(eu));
      }
    }
    *(unsigned*)(X + r * XLD + c) = val;
  }
  __syncthreads();

  mlp_gemm_relu<KD, XLD>(X, W1p, b1, Y1);
  __syncthreads();
  mlp_gemm_relu<128, kYLD>(Y1, W2p, b2, Y2);
  __syncthreads();
  mlp_gemm_sum(Y2, W3p, b3, mkA, colred, dh + (size_t)node * 128);
}

// ---------------------------------------------------------------------------
// Residual + LayerNorm (+ optional mask), writes f32 and bf16 mirrors.
// One 128-thread block per node row (H=128).
// ---------------------------------------------------------------------------
__global__ __launch_bounds__(128) void residual_ln_k(
    const float* __restrict__ x, const float* __restrict__ dh, float scale,
    const float* __restrict__ g, const float* __restrict__ bta,
    const float* __restrict__ maskV, float* __restrict__ outF,
    bf16_t* __restrict__ outB) {
  __shared__ float red[128];
  int row = blockIdx.x;
  int t = threadIdx.x;
  float v = x[(size_t)row * 128 + t] + (dh ? dh[(size_t)row * 128 + t] * scale : 0.f);
  red[t] = v;
  __syncthreads();
  for (int s = 64; s > 0; s >>= 1) { if (t < s) red[t] += red[t + s]; __syncthreads(); }
  float mu = red[0] * (1.f / 128.f);
  __syncthreads();
  float d = v - mu;
  red[t] = d * d;
  __syncthreads();
  for (int s = 64; s > 0; s >>= 1) { if (t < s) red[t] += red[t + s]; __syncthreads(); }
  float var = red[0] * (1.f / 128.f);
  float y = d * rsqrtf(var + 1e-6f) * g[t] + bta[t];
  if (maskV) y *= maskV[row];
  outF[(size_t)row * 128 + t] = y;
  outB[(size_t)row * 128 + t] = f2bf(y);
}

// ---------------------------------------------------------------------------
// Sequence embedding gather: hS[b,n,:] = W_s[S[b,n],:]  (bf16 out)
// ---------------------------------------------------------------------------
__global__ void embed_s_k(const int* __restrict__ S, const float* __restrict__ Ws,
                          bf16_t* __restrict__ hSb, int total) {
  int i = blockIdx.x * blockDim.x + threadIdx.x;
  if (i >= total * 128) return;
  int node = i >> 7;
  int c = i & 127;
  hSb[i] = f2bf(Ws[(size_t)S[node] * 128 + c]);
}

// ---------------------------------------------------------------------------
// Decoding order: stable argsort of chain_M*(|randn|+1e-3) per batch via
// rank counting, then scatter a[b, rank] = i.
// ---------------------------------------------------------------------------
__global__ void order_rank_k(const float* __restrict__ chain_M,
                             const float* __restrict__ randn,
                             int* __restrict__ a_out, int N, int total) {
  int idx = blockIdx.x * blockDim.x + threadIdx.x;
  if (idx >= total) return;
  int b = idx / N, i = idx % N;
  float vi = chain_M[idx] * (fabsf(randn[idx]) + 0.001f);
  int r = 0;
  for (int j = 0; j < N; ++j) {
    float vj = chain_M[(size_t)b * N + j] * (fabsf(randn[(size_t)b * N + j]) + 0.001f);
    r += (vj < vi) || (vj == vi && j < i);
  }
  a_out[(size_t)b * N + r] = i;
}

__global__ void mask_bw_k(const int* __restrict__ a, const int* __restrict__ E_idx,
                          const float* __restrict__ mask, float* __restrict__ mbw,
                          int N, int K, int total) {
  int e = blockIdx.x * blockDim.x + threadIdx.x;
  if (e >= total) return;
  int node = e / K;
  int b = node / N, i = node % N;
  int j = E_idx[e];
  float dec = (a[(size_t)b * N + i] > a[(size_t)b * N + j]) ? 1.f : 0.f;
  mbw[e] = mask[node] * dec;
}

// ---------------------------------------------------------------------------
// logits = hV @ W_out + b; log_softmax over 21.  One wave per node.
// ---------------------------------------------------------------------------
__global__ __launch_bounds__(32) void out_logits_k(const float* __restrict__ hV,
                                                   const float* __restrict__ Wout,
                                                   const float* __restrict__ bout,
                                                   float* __restrict__ out) {
  int node = blockIdx.x;
  int l = threadIdx.x;
  float acc = -INFINITY;
  if (l < 21) {
    float s = bout[l];
    for (int c = 0; c < 128; ++c) s += hV[(size_t)node * 128 + c] * Wout[c * 21 + l];
    acc = s;
  }
  float mx = acc;
  for (int off = 16; off > 0; off >>= 1) mx = fmaxf(mx, __shfl_xor(mx, off, 32));
  float ex = (l < 21) ? __expf(acc - mx) : 0.f;
  float sm = ex;
  for (int off = 16; off > 0; off >>= 1) sm += __shfl_xor(sm, off, 32);
  if (l < 21) out[(size_t)node * 21 + l] = acc - mx - __logf(sm);
}

// ---------------------------------------------------------------------------
// Host orchestration
// ---------------------------------------------------------------------------
extern "C" void kernel_launch(void* const* d_in, const int* in_sizes, int n_in,
                              void* d_out, int out_size, void* d_ws, size_t ws_size,
                              hipStream_t stream) {
  (void)in_sizes; (void)n_in; (void)out_size; (void)ws_size;
  constexpr int B = 8, N = 2048, KNB = 30, H = 128;
  constexpr int BN = B * N;
  constexpr size_t BNK = (size_t)BN * KNB;

  const float* V       = (const float*)d_in[0];
  const float* E       = (const float*)d_in[1];
  const int*   E_idx   = (const int*)d_in[2];
  const int*   S       = (const int*)d_in[3];
  const float* mask    = (const float*)d_in[4];
  const float* chain_M = (const float*)d_in[5];
  const float* randn   = (const float*)d_in[6];
  // params flattened as JAX pytree (dict keys sorted alphabetically):
  // W_e{b,w}=7,8  W_out{b,w}=9,10  W_s=11  W_v{b,w}=12,13
  // decoder layers at 14+14*i, encoder layers at 56+14*i; within a layer:
  // 0 W1.b 1 W1.w 2 W2.b 3 W2.w 4 W3.b 5 W3.w 6 d1.b 7 d1.w 8 d2.b 9 d2.w
  // 10 n1_b 11 n1_g 12 n2_b 13 n2_g
  const float* We_b   = (const float*)d_in[7];
  const float* We_w   = (const float*)d_in[8];
  const float* Wout_b = (const float*)d_in[9];
  const float* Wout_w = (const float*)d_in[10];
  const float* Ws     = (const float*)d_in[11];
  const float* Wv_b   = (const float*)d_in[12];
  const float* Wv_w   = (const float*)d_in[13];
  auto P = [&](int enc, int layer, int off) -> const float* {
    return (const float*)d_in[(enc ? 56 : 14) + 14 * layer + off];
  };

  // ---- workspace bump allocator -------------------------------------------
  char* wp = (char*)d_ws;
  auto bump = [&](size_t bytes) -> void* {
    void* r = (void*)wp;
    wp += (bytes + 255) & ~(size_t)255;
    return r;
  };
  float*  hV     = (float*)bump((size_t)BN * H * 4);
  float*  hln    = (float*)bump((size_t)BN * H * 4);
  float*  dh     = (float*)bump((size_t)BN * H * 4);
  float*  dh2    = (float*)bump((size_t)BN * H * 4);
  bf16_t* hVb    = (bf16_t*)bump((size_t)BN * H * 2);
  bf16_t* hlnb   = (bf16_t*)bump((size_t)BN * H * 2);
  bf16_t* hVencb = (bf16_t*)bump((size_t)BN * H * 2);
  bf16_t* hSb    = (bf16_t*)bump((size_t)BN * H * 2);
  bf16_t* y1b    = (bf16_t*)bump((size_t)BN * 4 * H * 2);
  bf16_t* hEb    = (bf16_t*)bump(BNK * H * 2);
  float*  mbw    = (float*)bump(BNK * 4);
  int*    aord   = (int*)bump((size_t)BN * 4);

  bf16_t* Wvp = (bf16_t*)bump(128 * 128 * 2);
  bf16_t* Wep = (bf16_t*)bump(128 * 128 * 2);
  bf16_t *eW1p[3], *eW2p[3], *eW3p[3], *ed1p[3], *ed2p[3];
  bf16_t *dW1p[3], *dW2p[3], *dW3p[3], *dd1p[3], *dd2p[3];
  for (int i = 0; i < 3; ++i) {
    eW1p[i] = (bf16_t*)bump(384 * 128 * 2);
    eW2p[i] = (bf16_t*)bump(128 * 128 * 2);
    eW3p[i] = (bf16_t*)bump(128 * 128 * 2);
    ed1p[i] = (bf16_t*)bump(128 * 512 * 2);
    ed2p[i] = (bf16_t*)bump(512 * 128 * 2);
    dW1p[i] = (bf16_t*)bump(512 * 128 * 2);
    dW2p[i] = (bf16_t*)bump(128 * 128 * 2);
    dW3p[i] = (bf16_t*)bump(128 * 128 * 2);
    dd1p[i] = (bf16_t*)bump(128 * 512 * 2);
    dd2p[i] = (bf16_t*)bump(512 * 128 * 2);
  }

  auto pack = [&](const float* W, bf16_t* Wpk, int K, int Npad) {
    int total = (K / 32) * (Npad / 16) * 256;
    pack_w_k<<<(total + 255) / 256, 256, 0, stream>>>(W, Wpk, K, Npad, Npad);
  };
  pack(Wv_w, Wvp, 128, 128);
  pack(We_w, Wep, 128, 128);
  for (int i = 0; i < 3; ++i) {
    pack(P(1, i, 1), eW1p[i], 384, 128);
    pack(P(1, i, 3), eW2p[i], 128, 128);
    pack(P(1, i, 5), eW3p[i], 128, 128);
    pack(P(1, i, 7), ed1p[i], 128, 512);
    pack(P(1, i, 9), ed2p[i], 512, 128);
    pack(P(0, i, 1), dW1p[i], 512, 128);
    pack(P(0, i, 3), dW2p[i], 128, 128);
    pack(P(0, i, 5), dW3p[i], 128, 128);
    pack(P(0, i, 7), dd1p[i], 128, 512);
    pack(P(0, i, 9), dd2p[i], 512, 128);
  }

  // ---- input embeddings ---------------------------------------------------
  gemm_k<float, 0, 1, 1><<<dim3(BN / 32, H / 16), 32, 0, stream>>>(
      V, Wvp, Wv_b, hV, hVb, H, H);
  gemm_k<float, 0, 0, 1><<<dim3((unsigned)(BNK / 32), H / 16), 32, 0, stream>>>(
      E, Wep, We_b, (float*)nullptr, hEb, H, H);

  // ---- encoder ------------------------------------------------------------
  for (int i = 0; i < 3; ++i) {
    edge_msg_k<0><<<BN, 128, 0, stream>>>(
        hVb, hEb, E_idx, mask, (const bf16_t*)nullptr, (const bf16_t*)nullptr,
        (const float*)nullptr, eW1p[i], P(1, i, 0), eW2p[i], P(1, i, 2),
        eW3p[i], P(1, i, 4), dh, N);
    residual_ln_k<<<BN, 128, 0, stream>>>(hV, dh, 1.f / 30.f, P(1, i, 11),
                                          P(1, i, 10), (const float*)nullptr, hln, hlnb);
    gemm_k<bf16_t, 1, 0, 1><<<dim3(BN / 32, 512 / 16), 32, 0, stream>>>(
        hlnb, ed1p[i], P(1, i, 6), (float*)nullptr, y1b, 128, 512);
    gemm_k<bf16_t, 0, 1, 0><<<dim3(BN / 32, 128 / 16), 32, 0, stream>>>(
        y1b, ed2p[i], P(1, i, 8), dh2, (bf16_t*)nullptr, 512, 128);
    residual_ln_k<<<BN, 128, 0, stream>>>(hln, dh2, 1.f, P(1, i, 13),
                                          P(1, i, 12), mask, hV, hVb);
  }

  // ---- decoder prep -------------------------------------------------------
  hipMemcpyAsync(hVencb, hVb, (size_t)BN * H * 2, hipMemcpyDeviceToDevice, stream);
  embed_s_k<<<(BN * H + 255) / 256, 256, 0, stream>>>(S, Ws, hSb, BN);
  order_rank_k<<<(BN + 255) / 256, 256, 0, stream>>>(chain_M, randn, aord, N, BN);
  mask_bw_k<<<(unsigned)((BNK + 255) / 256), 256, 0, stream>>>(
      aord, E_idx, mask, mbw, N, KNB, (int)BNK);

  // ---- decoder ------------------------------------------------------------
  for (int i = 0; i < 3; ++i) {
    edge_msg_k<1><<<BN, 128, 0, stream>>>(
        hVb, hEb, E_idx, mask, hSb, hVencb, mbw, dW1p[i], P(0, i, 0),
        dW2p[i], P(0, i, 2), dW3p[i], P(0, i, 4), dh, N);
    residual_ln_k<<<BN, 128, 0, stream>>>(hV, dh, 1.f / 30.f, P(0, i, 11),
                                          P(0, i, 10), (const float*)nullptr, hln, hlnb);
    gemm_k<bf16_t, 1, 0, 1><<<dim3(BN / 32, 512 / 16), 32, 0, stream>>>(
        hlnb, dd1p[i], P(0, i, 6), (float*)nullptr, y1b, 128, 512);
    gemm_k<bf16_t, 0, 1, 0><<<dim3(BN / 32, 128 / 16), 32, 0, stream>>>(
        y1b, dd2p[i], P(0, i, 8), dh2, (bf16_t*)nullptr, 512, 128);
    residual_ln_k<<<BN, 128, 0, stream>>>(hln, dh2, 1.f, P(0, i, 13),
                                          P(0, i, 12), mask, hV, hVb);
  }

  // ---- output head --------------------------------------------------------
  out_logits_k<<<BN, 32, 0, stream>>>(hV, Wout_w, Wout_b, (float*)d_out);
}
